// SCExtractorRL_51539607552892
// MI455X (gfx1250) — matrix operand order repaired and checked by reading, not compiled, not run on hardware
//
#include <hip/hip_runtime.h>
#include <hip/hip_bf16.h>

// ---------------------------------------------------------------------------
// SCExtractorRL decode for MI455X (gfx1250).
// Weight-streaming bound: bf16-convert + fragment-major swizzle all WMMA B
// operands once (prep), so the resident set (~160MB: LSTM weights 134MB,
// wq 16.8MB, bf16 feats 8.5MB, mem 4.3MB) fits the 192MB L2 and every
// WMMA B fragment is two contiguous global_load_b128 per lane.
// All matvecs/GEMM run through v_wmma_f32_16x16x32_bf16 (vector replicated
// across A rows for matvecs). Persistent kernel + atomic grid barrier carries
// the 1024 strictly-sequential decode steps.
// ---------------------------------------------------------------------------

typedef unsigned short u16;
typedef unsigned int   u32;
typedef __attribute__((ext_vector_type(16))) __bf16 v16bf;
typedef __attribute__((ext_vector_type(8)))  float  v8f;

#define NWG 128              // persistent workgroups (must be co-resident)
#define NEGV (-1.0e6f)
#define SMAX_SHIFT 16.0f     // scores are O(1); fixed-shift softmax is safe

struct Ctl {
  u32 bar_cnt;  u32 pad0[63];
  int stopped;  u32 pad1[63];
  int ext;      u32 pad2[63];
  float ssum;   u32 pad3[63];
};

union Frag { uint4 q[2]; u32 d[8]; v16bf v; };

struct P {
  // inputs (fp32)
  const float *attn_mem, *init_h, *init_c, *init_i, *stopv;
  const float *attn_wm, *attn_wq, *attn_v, *hop_wm, *hop_wq, *hop_v;
  const float *W_ih, *W_hh, *b_ih, *b_hh;
  // outputs
  float *outExt;   // [1024] extract indices (as float; -1 after stop)
  float *outS;     // [1024][1025] masked scores (0 after stop)
  // workspace
  Ctl  *ctl;
  int  *mask;      // [1040]
  float *c, *cNew; // [2][2048] fp32 cell state (+candidate)
  float *qw;       // [2048]  q @ wq result
  float *sbuf, *pbuf;          // [1040] scores / exp weights
  u16  *hB, *hNewB;            // [2][2048] bf16 hidden (+candidate)
  u16  *xB, *qB;               // [2048] bf16 lstm input / attention query
  u16  *memB;                  // [1040][2048] bf16 memory rows (+stop, pad)
  // fragment-major swizzled WMMA B operands (u32 = 2 packed bf16, k-pair):
  //   LSTM:  [(layer*512 + rowtile)*64 + kblk][lane][8]   (rowtile=gate*128+wg)
  //   wq/wm: [kblk*128 + ntile][lane][8]
  u32  *wihS, *whhS;           // 2 * 2^23 u32 each
  u32  *attnWqS, *hopWqS, *attnWmS, *hopWmS;   // 2^21 u32 each
  u16  *attnFeatB, *hopFeatB;  // [1040][2048] bf16 features
};

__device__ __forceinline__ u16 f2b(float x) {      // fp32 -> bf16 RNE
  u32 u = __float_as_uint(x);
  return (u16)((u + 0x7fffu + ((u >> 16) & 1u)) >> 16);
}
__device__ __forceinline__ float b2f(u16 b) {
  return __uint_as_float((u32)b << 16);
}
__device__ __forceinline__ u32 packpair(float lo, float hi) {
  return ((u32)f2b(hi) << 16) | (u32)f2b(lo);
}
__device__ __forceinline__ float sigm(float x) { return 1.0f / (1.0f + __expf(-x)); }

__device__ __forceinline__ v8f bf16_wmma(const Frag& a, const Frag& b, v8f c) {
  return __builtin_amdgcn_wmma_f32_16x16x32_bf16(false, a.v, false, b.v,
                                                 (short)0, c, false, false);
}

__device__ __forceinline__ void grid_sync(Ctl* ctl) {
  __syncthreads();
  if (threadIdx.x == 0) {
    __threadfence();
    u32 ticket = __hip_atomic_fetch_add(&ctl->bar_cnt, 1u,
                                        __ATOMIC_ACQ_REL, __HIP_MEMORY_SCOPE_AGENT);
    u32 target = (ticket / NWG + 1u) * NWG;
    while (__hip_atomic_load(&ctl->bar_cnt, __ATOMIC_ACQUIRE,
                             __HIP_MEMORY_SCOPE_AGENT) < target)
      __builtin_amdgcn_s_sleep(1);
  }
  __syncthreads();
}

// ---------------------------------------------------------------------------
// Kernel 1: conversions + swizzles + state init (grid-stride, deterministic)
// ---------------------------------------------------------------------------
__global__ void prep_kernel(P p) {
  size_t gid = blockIdx.x * (size_t)blockDim.x + threadIdx.x;
  size_t gst = gridDim.x * (size_t)blockDim.x;
  if (gid == 0) { p.ctl->bar_cnt = 0; p.ctl->stopped = 0; p.ctl->ext = 0; p.ctl->ssum = 0.f; }
  for (size_t i = gid; i < 1040; i += gst) p.mask[i] = 0;
  for (size_t i = gid; i < 2 * 2048; i += gst) {
    p.c[i] = p.init_c[i]; p.hB[i] = f2b(p.init_h[i]);
  }
  for (size_t i = gid; i < 2048; i += gst) p.xB[i] = f2b(p.init_i[i]);
  for (size_t i = gid; i < (size_t)1040 * 2048; i += gst) {
    size_t r = i >> 11, k = i & 2047;
    float v = (r < 1024) ? p.attn_mem[i] : (r == 1024 ? p.stopv[k] : 0.f);
    p.memB[i] = f2b(v);
  }
  // LSTM weights -> fragment-major swizzle (2^24 u32 per array)
  for (size_t i = gid; i < ((size_t)1 << 24); i += gst) {
    int v    = (int)(i & 7);
    int lane = (int)((i >> 3) & 31);
    int blk  = (int)((i >> 8) & 63);
    int rt   = (int)((i >> 14) & 511);
    int lyr  = (int)((i >> 23) & 1);
    size_t row = (size_t)lyr * 8192 + (size_t)rt * 16 + (lane & 15);
    int k = blk * 32 + (v >> 2) * 16 + (lane >> 4) * 8 + 2 * (v & 3);
    size_t s = row * 2048 + k;
    p.wihS[i] = packpair(p.W_ih[s], p.W_ih[s + 1]);
    p.whhS[i] = packpair(p.W_hh[s], p.W_hh[s + 1]);
  }
  // wq / wm -> fragment-major swizzle (2^21 u32 per array)
  for (size_t i = gid; i < ((size_t)1 << 21); i += gst) {
    int v    = (int)(i & 7);
    int lane = (int)((i >> 3) & 31);
    int nt   = (int)((i >> 8) & 127);
    int blk  = (int)((i >> 15) & 63);
    int col  = nt * 16 + (lane & 15);
    int k    = blk * 32 + (v >> 2) * 16 + (lane >> 4) * 8 + 2 * (v & 3);
    size_t s0 = (size_t)k * 2048 + col, s1 = (size_t)(k + 1) * 2048 + col;
    p.attnWqS[i] = packpair(p.attn_wq[s0], p.attn_wq[s1]);
    p.hopWqS[i]  = packpair(p.hop_wq[s0],  p.hop_wq[s1]);
    p.attnWmS[i] = packpair(p.attn_wm[s0], p.attn_wm[s1]);
    p.hopWmS[i]  = packpair(p.hop_wm[s0],  p.hop_wm[s1]);
  }
}

// ---------------------------------------------------------------------------
// Kernel 2: feats = mem @ wm (both), WMMA bf16 GEMM, one 16x16 tile per wave.
// jobs = 2 * 65 Mtiles * 128 Ntiles = 16640 waves = 2080 blocks of 8 waves.
// ---------------------------------------------------------------------------
__global__ void feat_gemm_kernel(P p) {
  int wgl  = blockIdx.x * 8 + (threadIdx.x >> 5);
  int lane = threadIdx.x & 31;
  int m = lane & 15, half = lane >> 4;
  int which = (wgl >= 65 * 128);
  int rem = which ? wgl - 65 * 128 : wgl;
  int mt = rem >> 7, nt = rem & 127;
  const u32* WS = which ? p.hopWmS : p.attnWmS;
  u16*       F  = which ? p.hopFeatB : p.attnFeatB;
  const u16* arow = p.memB + (size_t)(mt * 16 + m) * 2048 + half * 8;
  int coln = nt * 16 + m;
  v8f acc = {0.f, 0.f, 0.f, 0.f, 0.f, 0.f, 0.f, 0.f};
  for (int kc = 0; kc < 2048; kc += 32) {
    int blk = kc >> 5;
    const u32* wb = WS + ((((size_t)blk * 128 + nt) * 32 + lane) << 3);
    Frag a, b;
    a.q[0] = *(const uint4*)(arow + kc);
    a.q[1] = *(const uint4*)(arow + kc + 16);
    b.q[0] = *(const uint4*)(wb);
    b.q[1] = *(const uint4*)(wb + 4);
    acc = bf16_wmma(a, b, acc);
  }
#pragma unroll
  for (int v = 0; v < 8; ++v)
    F[(size_t)(mt * 16 + v + half * 8) * 2048 + coln] = f2b(acc[v]);
}

// ---------------------------------------------------------------------------
// Kernel 3: persistent sequential decode.
// ---------------------------------------------------------------------------
__device__ __forceinline__ void lstm_layer(
    const u16* __restrict__ x, const u16* __restrict__ hB,
    const u32* __restrict__ WihS, const u32* __restrict__ WhhS,  // layer base
    const float* __restrict__ bih, const float* __restrict__ bhh,
    const float* __restrict__ cOld, float* __restrict__ cNew,
    u16* __restrict__ hNew, float (*red)[16], int wg)
{
  const int tid = threadIdx.x, wv = tid >> 5, lane = tid & 31;
  const int gate = wv >> 1, part = wv & 1;
  const int half = lane >> 4;
  const u16* vec = part ? hB : x;
  const u32* WS  = part ? WhhS : WihS;
  const int rt = gate * 128 + wg;                    // row tile 0..511
  const u32* wbase = WS + (((size_t)rt * 64 * 32 + lane) << 3);
  v8f acc = {0.f, 0.f, 0.f, 0.f, 0.f, 0.f, 0.f, 0.f};
  for (int kc = 0; kc < 2048; kc += 32) {
    const u32* wb = wbase + (size_t)(kc >> 5) * 256;   // 32 lanes * 8 dwords
    __builtin_prefetch(wb + 512, 0, 1);                // stream 2 chunks ahead
    Frag a, b;
    a.q[0] = *(const uint4*)(vec + kc + half * 8);
    a.q[1] = *(const uint4*)(vec + kc + 16 + half * 8);
    b.q[0] = *(const uint4*)(wb);
    b.q[1] = *(const uint4*)(wb + 4);
    acc = bf16_wmma(a, b, acc);
  }
  if (lane < 16) red[wv][lane] = acc[0];   // replicated-A => row 0 = matvec
  __syncthreads();
  if (tid < 16) {
    int j = wg * 16 + tid;
    float gi = red[0][tid] + red[1][tid] + bih[0 * 2048 + j] + bhh[0 * 2048 + j];
    float gf = red[2][tid] + red[3][tid] + bih[1 * 2048 + j] + bhh[1 * 2048 + j];
    float gg = red[4][tid] + red[5][tid] + bih[2 * 2048 + j] + bhh[2 * 2048 + j];
    float go = red[6][tid] + red[7][tid] + bih[3 * 2048 + j] + bhh[3 * 2048 + j];
    float c1 = sigm(gf) * cOld[j] + sigm(gi) * tanhf(gg);
    cNew[j] = c1;
    hNew[j] = f2b(sigm(go) * tanhf(c1));
  }
  __syncthreads();
}

__device__ __forceinline__ void matvec2048(
    const u16* __restrict__ vec, const u32* __restrict__ WS,
    float* __restrict__ out, float (*red)[16], int wg, bool resetSum, Ctl* ctl)
{
  const int tid = threadIdx.x, wv = tid >> 5, lane = tid & 31;
  const int half = lane >> 4;
  v8f acc = {0.f, 0.f, 0.f, 0.f, 0.f, 0.f, 0.f, 0.f};
  const int kb = wv * 256;
  for (int kc = kb; kc < kb + 256; kc += 32) {
    int blk = kc >> 5;
    const u32* wb = WS + ((((size_t)blk * 128 + wg) * 32 + lane) << 3);
    Frag a, b;
    a.q[0] = *(const uint4*)(vec + kc + half * 8);
    a.q[1] = *(const uint4*)(vec + kc + 16 + half * 8);
    b.q[0] = *(const uint4*)(wb);
    b.q[1] = *(const uint4*)(wb + 4);
    acc = bf16_wmma(a, b, acc);
  }
  if (lane < 16) red[wv][lane] = acc[0];
  if (resetSum && blockIdx.x == 0 && tid == 0) ctl->ssum = 0.f; // safe: pre-sync
  __syncthreads();
  if (tid < 16) {
    float s = 0.f;
#pragma unroll
    for (int w = 0; w < 8; ++w) s += red[w][tid];
    out[wg * 16 + tid] = s;
  }
  __syncthreads();
}

__global__ void decode_kernel(P p) {
  __shared__ float red[8][16];
  __shared__ float cpart[16][16];
  __shared__ float bv[256];
  __shared__ int   bi[256];
  __shared__ int   sTop0;
  const int tid = threadIdx.x, wg = blockIdx.x;
  const int wv = tid >> 5, lane = tid & 31;
  const size_t L1S = (size_t)1 << 23;       // layer stride in swizzled u32
  int tStop = 1024;

  for (int t = 0; t < 1024; ++t) {
    // --- LSTM layer 0 ---
    lstm_layer(p.xB, p.hB, p.wihS, p.whhS, p.b_ih, p.b_hh,
               p.c, p.cNew, p.hNewB, red, wg);
    grid_sync(p.ctl);
    // --- LSTM layer 1 (query source) ---
    lstm_layer(p.hNewB, p.hB + 2048, p.wihS + L1S, p.whhS + L1S,
               p.b_ih + 8192, p.b_hh + 8192,
               p.c + 2048, p.cNew + 2048, p.hNewB + 2048, red, wg);
    if (tid < 16) p.qB[wg * 16 + tid] = p.hNewB[2048 + wg * 16 + tid];
    grid_sync(p.ctl);

    // --- attention hops ---
    for (int hop = 0; hop < 2; ++hop) {
      matvec2048(p.qB, p.hopWqS, p.qw, red, wg, /*resetSum=*/true, p.ctl);
      grid_sync(p.ctl);
      // scores + exp + sum (one wave per row; 8 contiguous elems per lane)
      for (int r = wg * 8 + wv; r < 1025; r += NWG * 8) {
        const u16* fr = p.hopFeatB + (size_t)r * 2048;
        float a = 0.f;
        for (int k0 = lane * 8; k0 < 2048; k0 += 256) {
#pragma unroll
          for (int j = 0; j < 8; ++j)
            a += tanhf(b2f(fr[k0 + j]) + p.qw[k0 + j]) * p.hop_v[k0 + j];
        }
        for (int off = 16; off; off >>= 1) a += __shfl_xor(a, off, 32);
        if (lane == 0) {
          float e = __expf(a - SMAX_SHIFT);
          p.pbuf[r] = e;
          atomicAdd(&p.ctl->ssum, e);
        }
      }
      grid_sync(p.ctl);
      // context: q = softmax(s) @ hop_feat   (16 cols per WG)
      {
        int cl = tid & 15, rg = tid >> 4;
        int col = wg * 16 + cl;
        float partial = 0.f;
        for (int r = rg; r < 1025; r += 16)
          partial += p.pbuf[r] * b2f(p.hopFeatB[(size_t)r * 2048 + col]);
        cpart[rg][cl] = partial;
        __syncthreads();
        if (tid < 16) {
          float s = 0.f;
#pragma unroll
          for (int g = 0; g < 16; ++g) s += cpart[g][tid];
          p.qB[wg * 16 + tid] = f2b(s / p.ctl->ssum);
        }
        __syncthreads();
      }
      grid_sync(p.ctl);
    }

    // --- final attention scores (masked) ---
    matvec2048(p.qB, p.attnWqS, p.qw, red, wg, /*resetSum=*/false, p.ctl);
    grid_sync(p.ctl);
    for (int r = wg * 8 + wv; r < 1025; r += NWG * 8) {
      const u16* fr = p.attnFeatB + (size_t)r * 2048;
      float a = 0.f;
      for (int k0 = lane * 8; k0 < 2048; k0 += 256) {
#pragma unroll
        for (int j = 0; j < 8; ++j)
          a += tanhf(b2f(fr[k0 + j]) + p.qw[k0 + j]) * p.attn_v[k0 + j];
      }
      for (int off = 16; off; off >>= 1) a += __shfl_xor(a, off, 32);
      if (lane == 0) {
        float m = p.mask[r] ? NEGV : a;
        p.sbuf[r] = m;
        p.outS[(size_t)t * 1025 + r] = m;
      }
    }
    grid_sync(p.ctl);

    // --- top-2 + control (WG0) ---
    if (wg == 0) {
      float best = -3.4e38f; int bidx = 1025;
      for (int r = tid; r < 1025; r += 256) {
        float v = p.sbuf[r];
        if (v > best) { best = v; bidx = r; }
      }
      bv[tid] = best; bi[tid] = bidx; __syncthreads();
      for (int s = 128; s > 0; s >>= 1) {
        if (tid < s) {
          float v2 = bv[tid + s]; int i2 = bi[tid + s];
          if (v2 > bv[tid] || (v2 == bv[tid] && i2 < bi[tid])) { bv[tid] = v2; bi[tid] = i2; }
        }
        __syncthreads();
      }
      if (tid == 0) sTop0 = bi[0];
      __syncthreads();
      best = -3.4e38f; bidx = 1025;
      for (int r = tid; r < 1025; r += 256) {
        if (r == sTop0) continue;
        float v = p.sbuf[r];
        if (v > best) { best = v; bidx = r; }
      }
      bv[tid] = best; bi[tid] = bidx; __syncthreads();
      for (int s = 128; s > 0; s >>= 1) {
        if (tid < s) {
          float v2 = bv[tid + s]; int i2 = bi[tid + s];
          if (v2 > bv[tid] || (v2 == bv[tid] && i2 < bi[tid])) { bv[tid] = v2; bi[tid] = i2; }
        }
        __syncthreads();
      }
      if (tid == 0) {
        int t0 = sTop0, t1 = bi[0];
        bool is_stop = (t0 == 1024);
        int ext = (t == 0 && is_stop) ? t1 : t0;
        bool ns = (t != 0) && is_stop;
        if (ns) { p.ctl->stopped = 1; p.outExt[t] = -1.f; }
        else    { p.ctl->ext = ext; p.mask[ext] = 1; p.outExt[t] = (float)ext; }
      }
    }
    grid_sync(p.ctl);

    if (p.ctl->stopped) { tStop = t; break; }

    // --- commit state + gather next LSTM input ---
    {
      int ext = p.ctl->ext;
      int g = wg * 256 + tid;                 // 0..32767
      if (g < 4096) { p.hB[g] = p.hNewB[g]; p.c[g] = p.cNew[g]; }
      else if (g < 6144) {
        int k = g - 4096;
        p.xB[k] = p.memB[(size_t)ext * 2048 + k];
      }
    }
    grid_sync(p.ctl);
  }

  // --- fill after stop: scores rows = 0.0, extracts = -1 ---
  if (tStop < 1024) {
    size_t gid = wg * (size_t)256 + tid, gst = (size_t)NWG * 256;
    size_t total = (size_t)(1024 - tStop) * 1025;
    float* base = p.outS + (size_t)tStop * 1025;
    for (size_t i = gid; i < total; i += gst) base[i] = 0.f;
    for (size_t i = tStop + gid; i < 1024; i += gst) p.outExt[i] = -1.f;
  }
}

// ---------------------------------------------------------------------------
extern "C" void kernel_launch(void* const* d_in, const int* in_sizes, int n_in,
                              void* d_out, int out_size, void* d_ws, size_t ws_size,
                              hipStream_t stream) {
  (void)in_sizes; (void)n_in; (void)out_size; (void)ws_size;
  P p;
  p.attn_mem = (const float*)d_in[0];
  p.init_h   = (const float*)d_in[1];
  p.init_c   = (const float*)d_in[2];
  p.init_i   = (const float*)d_in[3];
  p.stopv    = (const float*)d_in[4];
  p.attn_wm  = (const float*)d_in[5];
  p.attn_wq  = (const float*)d_in[6];
  p.attn_v   = (const float*)d_in[7];
  p.hop_wm   = (const float*)d_in[8];
  p.hop_wq   = (const float*)d_in[9];
  p.hop_v    = (const float*)d_in[10];
  p.W_ih     = (const float*)d_in[11];
  p.W_hh     = (const float*)d_in[12];
  p.b_ih     = (const float*)d_in[13];
  p.b_hh     = (const float*)d_in[14];

  p.outExt = (float*)d_out;
  p.outS   = (float*)d_out + 1024;

  size_t off = 0;
  auto take = [&](size_t bytes) {
    char* ptr = (char*)d_ws + off;
    off += (bytes + 255) & ~(size_t)255;
    return (void*)ptr;
  };
  p.ctl       = (Ctl*)  take(sizeof(Ctl));
  p.mask      = (int*)  take(1040 * 4);
  p.c         = (float*)take(2 * 2048 * 4);
  p.cNew      = (float*)take(2 * 2048 * 4);
  p.qw        = (float*)take(2048 * 4);
  p.sbuf      = (float*)take(1040 * 4);
  p.pbuf      = (float*)take(1040 * 4);
  p.hB        = (u16*)  take(2 * 2048 * 2);
  p.hNewB     = (u16*)  take(2 * 2048 * 2);
  p.xB        = (u16*)  take(2048 * 2);
  p.qB        = (u16*)  take(2048 * 2);
  p.memB      = (u16*)  take((size_t)1040 * 2048 * 2);
  p.attnFeatB = (u16*)  take((size_t)1040 * 2048 * 2);
  p.hopFeatB  = (u16*)  take((size_t)1040 * 2048 * 2);
  p.wihS      = (u32*)  take(((size_t)1 << 24) * 4);
  p.whhS      = (u32*)  take(((size_t)1 << 24) * 4);
  p.attnWqS   = (u32*)  take(((size_t)1 << 21) * 4);
  p.hopWqS    = (u32*)  take(((size_t)1 << 21) * 4);
  p.attnWmS   = (u32*)  take(((size_t)1 << 21) * 4);
  p.hopWmS    = (u32*)  take(((size_t)1 << 21) * 4);

  prep_kernel<<<2048, 256, 0, stream>>>(p);
  feat_gemm_kernel<<<2080, 256, 0, stream>>>(p);
  decode_kernel<<<NWG, 256, 0, stream>>>(p);
}
